// LGNNLayer_76579266887989
// MI455X (gfx1250) — matrix operand order repaired and checked by reading, not compiled, not run on hardware
//
#include <hip/hip_runtime.h>
#include <hip/hip_bf16.h>

#define DD  128   // feature dim
#define TM  64    // rows per GEMM block tile (E % TM == 0 for this problem: 80000/64=1250)
#define SAP 136   // padded LDS pitch (elements) -> rows land on different banks

typedef __attribute__((ext_vector_type(16))) __bf16 v16bf;
typedef __attribute__((ext_vector_type(8)))  __bf16 v8bf;
typedef __attribute__((ext_vector_type(8)))  float  v8f;

__device__ __forceinline__ __bf16 f32_to_bf16(float f) {
  union { float f; unsigned u; } in; in.f = f;
  unsigned u = in.u;
  u += 0x7FFFu + ((u >> 16) & 1u);           // round-to-nearest-even
  union { unsigned short s; __bf16 b; } out;
  out.s = (unsigned short)(u >> 16);
  return out.b;
}

// out = act(A[rows,128] @ W[128,128] + bias), act = ReLU if do_relu.
// One block = 128 threads = 4 wave32; each wave computes a 16x128 slice via
// v_wmma_f32_16x16x32_bf16 (8 N-tiles x 4 K-chunks). In-place (out==A) is safe:
// the A tile is staged to LDS before any store.
__global__ __launch_bounds__(128) void lgnn_gemm128(
    const float* __restrict__ A, const float* __restrict__ W,
    const float* __restrict__ bias, float* __restrict__ outp, int do_relu)
{
  __shared__ __bf16 sA[TM * SAP];   // A tile, bf16, padded pitch
  __shared__ __bf16 sW[DD * SAP];   // W transposed: sW[n*SAP+k] = W[k*DD+n]
  __shared__ float  sB[DD];

  const int  tid     = threadIdx.x;            // 0..127
  const long rowBase = (long)blockIdx.x * TM;

  // ---- stage A tile (TM*DD f32 = 2048 float4; 16 per thread) ----
  {
    const float4* Ag = (const float4*)(A + rowBase * DD);
#pragma unroll
    for (int it = 0; it < 16; ++it) {
      int q = it * 128 + tid;                  // float4 index within tile
      float4 v = Ag[q];
      int e = q * 4;
      int r = e >> 7;                          // row (float4 never crosses rows)
      int d = e & 127;
      __bf16* p = &sA[r * SAP + d];
      p[0] = f32_to_bf16(v.x); p[1] = f32_to_bf16(v.y);
      p[2] = f32_to_bf16(v.z); p[3] = f32_to_bf16(v.w);
    }
  }
  // ---- stage W transposed (strided global reads; k staggered over banks) ----
  {
    const int n = tid;
    for (int k0 = 0; k0 < DD; ++k0) {
      int k = (k0 + 2 * tid) & 127;
      sW[n * SAP + k] = f32_to_bf16(W[k * DD + n]);
    }
    sB[tid] = bias[tid];
  }
  __syncthreads();

  const int lane  = tid & 31;
  const int wave  = tid >> 5;
  const int mrow  = lane & 15;
  const int khalf = lane >> 4;                 // 0 or 1
  const int rbase = wave * 16;

  v8f acc[8];
#pragma unroll
  for (int i = 0; i < 8; ++i)
#pragma unroll
    for (int j = 0; j < 8; ++j) acc[i][j] = 0.0f;

#pragma unroll
  for (int c = 0; c < 4; ++c) {                // K chunks of 32
    // A fragment (contiguous K order): lanes 0-15 K=kb..kb+15, lanes 16-31 K=kb+16..kb+31
    const int kb = c * 32 + khalf * 16;
    const __bf16* ap = &sA[(rbase + mrow) * SAP + kb];
    v8bf alo = *(const v8bf*)ap;
    v8bf ahi = *(const v8bf*)(ap + 8);
    v16bf afrag;
#pragma unroll
    for (int i = 0; i < 8; ++i) { afrag[i] = alo[i]; afrag[i + 8] = ahi[i]; }
#pragma unroll
    for (int nt = 0; nt < 8; ++nt) {           // N tiles of 16
      // B fragment: column N=lane&15 of W (rows of sW), K-half by lane>>4
      const __bf16* bp = &sW[(nt * 16 + mrow) * SAP + c * 32 + khalf * 16];
      v8bf blo = *(const v8bf*)bp;
      v8bf bhi = *(const v8bf*)(bp + 8);
      v16bf bfrag;
#pragma unroll
      for (int i = 0; i < 8; ++i) { bfrag[i] = blo[i]; bfrag[i + 8] = bhi[i]; }
      acc[nt] = __builtin_amdgcn_wmma_f32_16x16x32_bf16(
          false, afrag, false, bfrag, (short)0, acc[nt], false, false);
    }
  }

  // ---- epilogue: bias + optional ReLU, C/D layout M = r + 8*khalf ----
#pragma unroll
  for (int nt = 0; nt < 8; ++nt) {
    const int col = nt * 16 + mrow;
    const float bv = sB[col];
#pragma unroll
    for (int r = 0; r < 8; ++r) {
      float v = acc[nt][r] + bv;
      if (do_relu) v = fmaxf(v, 0.0f);
      outp[(rowBase + rbase + khalf * 8 + r) * DD + col] = v;
    }
  }
}

// lgX[e] = 0.5*(x[src[e]] + x[dst[e]])
__global__ void lgnn_init_lgx(const float* __restrict__ x, const int* __restrict__ src,
                              const int* __restrict__ dst, float* __restrict__ lgX, int E)
{
  int t = blockIdx.x * blockDim.x + threadIdx.x;
  if (t >= E * (DD / 4)) return;
  int i = t >> 5;
  int d = (t & 31) << 2;
  const float4 a = *(const float4*)&x[(long)src[i] * DD + d];
  const float4 b = *(const float4*)&x[(long)dst[i] * DD + d];
  float4 r;
  r.x = 0.5f * (a.x + b.x); r.y = 0.5f * (a.y + b.y);
  r.z = 0.5f * (a.z + b.z); r.w = 0.5f * (a.w + b.w);
  *(float4*)&lgX[(long)i * DD + d] = r;
}

__global__ void lgnn_zero(float* __restrict__ p, int n) {
  int t = blockIdx.x * blockDim.x + threadIdx.x;
  if (t < n) p[t] = 0.0f;
}

// S[dst[e]] += relu(lgX[e] + x[dst[e]])   (collapsed L-edge segment_sum)
__global__ void lgnn_scatter_msg(const float* __restrict__ lgX, const float* __restrict__ x,
                                 const int* __restrict__ dst, float* __restrict__ S, int E)
{
  int t = blockIdx.x * blockDim.x + threadIdx.x;
  if (t >= E * (DD / 4)) return;
  int i = t >> 5;
  int d = (t & 31) << 2;
  int v = dst[i];
  const float4 a = *(const float4*)&lgX[(long)i * DD + d];
  const float4 b = *(const float4*)&x[(long)v * DD + d];
  float* Sp = &S[(long)v * DD + d];
  atomicAdd(Sp + 0, fmaxf(a.x + b.x, 0.0f));
  atomicAdd(Sp + 1, fmaxf(a.y + b.y, 0.0f));
  atomicAdd(Sp + 2, fmaxf(a.z + b.z, 0.0f));
  atomicAdd(Sp + 3, fmaxf(a.w + b.w, 0.0f));
}

// h[e] = lgX[e] + S[src[e]]
__global__ void lgnn_add_aggr(const float* __restrict__ lgX, const float* __restrict__ S,
                              const int* __restrict__ src, float* __restrict__ h, int E)
{
  int t = blockIdx.x * blockDim.x + threadIdx.x;
  if (t >= E * (DD / 4)) return;
  int i = t >> 5;
  int d = (t & 31) << 2;
  int v = src[i];
  const float4 a = *(const float4*)&lgX[(long)i * DD + d];
  const float4 s = *(const float4*)&S[(long)v * DD + d];
  float4 r; r.x = a.x + s.x; r.y = a.y + s.y; r.z = a.z + s.z; r.w = a.w + s.w;
  *(float4*)&h[(long)i * DD + d] = r;
}

// S[dst[e]] += lgX[e] ; cnt[dst[e]] += 1
__global__ void lgnn_scatter_out(const float* __restrict__ lgX, const int* __restrict__ dst,
                                 float* __restrict__ S, float* __restrict__ cnt, int E)
{
  int t = blockIdx.x * blockDim.x + threadIdx.x;
  if (t >= E * (DD / 4)) return;
  int i = t >> 5;
  int d = (t & 31) << 2;
  int v = dst[i];
  const float4 a = *(const float4*)&lgX[(long)i * DD + d];
  float* Sp = &S[(long)v * DD + d];
  atomicAdd(Sp + 0, a.x);
  atomicAdd(Sp + 1, a.y);
  atomicAdd(Sp + 2, a.z);
  atomicAdd(Sp + 3, a.w);
  if ((t & 31) == 0) atomicAdd(&cnt[v], 1.0f);
}

// out[v] = relu(S[v] / max(cnt[v],1))
__global__ void lgnn_finalize(const float* __restrict__ S, const float* __restrict__ cnt,
                              float* __restrict__ outp, int Nn)
{
  int t = blockIdx.x * blockDim.x + threadIdx.x;
  if (t >= Nn * (DD / 4)) return;
  int v = t >> 5;
  int d = (t & 31) << 2;
  float inv = 1.0f / fmaxf(cnt[v], 1.0f);
  const float4 s = *(const float4*)&S[(long)v * DD + d];
  float4 r;
  r.x = fmaxf(s.x * inv, 0.0f); r.y = fmaxf(s.y * inv, 0.0f);
  r.z = fmaxf(s.z * inv, 0.0f); r.w = fmaxf(s.w * inv, 0.0f);
  *(float4*)&outp[(long)v * DD + d] = r;
}

extern "C" void kernel_launch(void* const* d_in, const int* in_sizes, int n_in,
                              void* d_out, int out_size, void* d_ws, size_t ws_size,
                              hipStream_t stream)
{
  const float* x  = (const float*)d_in[0];
  const int*   ei = (const int*)d_in[1];   // [2,E]
  // d_in[2]=lg_src, d_in[3]=lg_dst: implied by structure, not needed
  const float* W1 = (const float*)d_in[4];
  const float* b1 = (const float*)d_in[5];
  const float* W2 = (const float*)d_in[6];
  const float* b2 = (const float*)d_in[7];
  float* outp = (float*)d_out;
  (void)n_in; (void)out_size; (void)ws_size;

  const int Nn = in_sizes[0] / DD;
  const int E  = in_sizes[1] / 2;
  const int* src = ei;
  const int* dst = ei + E;

  float* lgX = (float*)d_ws;                 // [E, D]
  float* h   = lgX + (size_t)E * DD;         // [E, D]  (aliased as T in-place)
  float* S   = h   + (size_t)E * DD;         // [N, D]
  float* cnt = S   + (size_t)Nn * DD;        // [N]     (contiguous after S)

  const int TPB = 256;
  const int gE  = (E * (DD / 4) + TPB - 1) / TPB;
  const int gN  = (Nn * (DD / 4) + TPB - 1) / TPB;
  const int nS  = Nn * DD;
  const int gZ  = (nS + TPB - 1) / TPB;
  const int gZ2 = (nS + Nn + TPB - 1) / TPB;
  const int gG  = E / TM;                    // 1250 for E=80000

  lgnn_init_lgx<<<gE, TPB, 0, stream>>>(x, src, dst, lgX, E);

  for (int it = 0; it < 3; ++it) {
    lgnn_zero<<<gZ, TPB, 0, stream>>>(S, nS);
    lgnn_scatter_msg<<<gE, TPB, 0, stream>>>(lgX, x, dst, S, E);
    lgnn_add_aggr<<<gE, TPB, 0, stream>>>(lgX, S, src, h, E);
    lgnn_gemm128<<<gG, 128, 0, stream>>>(h, W1, b1, h, 1);    // T = relu(h@W1+b1), in-place
    lgnn_gemm128<<<gG, 128, 0, stream>>>(h, W2, b2, lgX, 0);  // lgX = T@W2+b2
  }

  lgnn_zero<<<gZ2, TPB, 0, stream>>>(S, nS + Nn);             // zeros S and cnt
  lgnn_scatter_out<<<gE, TPB, 0, stream>>>(lgX, dst, S, cnt, E);
  lgnn_finalize<<<gN, TPB, 0, stream>>>(S, cnt, outp, Nn);
}
